// MultiHeadSelfAttention_37546604102087
// MI455X (gfx1250) — matrix operand order repaired and checked by reading, not compiled, not run on hardware
//
#include <hip/hip_runtime.h>
#include <hip/hip_bf16.h>
#include <math.h>
#include <stdint.h>

typedef unsigned short u16;
typedef __attribute__((ext_vector_type(16))) __bf16 v16bf;
typedef __attribute__((ext_vector_type(8)))  float  v8f;
typedef __attribute__((ext_vector_type(8)))  unsigned int v8u;
typedef __attribute__((ext_vector_type(4)))  int v4i;

#define BATCH 2
#define SEQ   2048
#define DMODEL 1024
#define HEADS 16
#define DK    64
#define MROWS (BATCH * SEQ)          // 4096

// ---------------- async global->LDS (CDNA5 ASYNCcnt path), guarded ----------------
#if defined(__AMDGCN__) && __has_builtin(__builtin_amdgcn_global_load_async_to_lds_b128) && \
    __has_builtin(__builtin_amdgcn_s_wait_asynccnt)
#define USE_ASYNC_LDS 1
#else
#define USE_ASYNC_LDS 0
#endif

__device__ __forceinline__ void copy16_g2lds(u16* lds_dst, const u16* gsrc) {
#if USE_ASYNC_LDS
  __builtin_amdgcn_global_load_async_to_lds_b128((v4i*)gsrc, (v4i*)lds_dst, 0, 0);
#else
  *(uint4*)lds_dst = *(const uint4*)gsrc;
#endif
}

__device__ __forceinline__ void wait_async_copies() {
#if USE_ASYNC_LDS
  __builtin_amdgcn_s_wait_asynccnt(0);
#endif
}

// ---------------- scalar helpers ----------------

__device__ __forceinline__ u16 f2bf(float f) {
  unsigned u = __builtin_bit_cast(unsigned, f);
  u += 0x7FFFu + ((u >> 16) & 1u);   // round-to-nearest-even
  return (u16)(u >> 16);
}
__device__ __forceinline__ unsigned pack_bf16(float a, float b) {
  return (unsigned)f2bf(a) | ((unsigned)f2bf(b) << 16);
}

// ---------------- WMMA fragment loaders (CDNA5 16x16x32 bf16 layouts) ----------------

// A fragment (16x32 bf16, row-major, leading dim ld).
// lane<16: row=lane, K {0..7,16..23}; lane>=16: row=lane-16, K {8..15,24..31}
__device__ __forceinline__ v16bf load_frag_a_bf16(const u16* tile, int ld) {
  const int lane = threadIdx.x & 31;
  const int r = lane & 15;
  const int c = (lane < 16) ? 0 : 8;
  const u16* p = tile + r * ld + c;
  uint4 lo = *(const uint4*)(p);
  uint4 hi = *(const uint4*)(p + 16);
  v8u u;
  u[0] = lo.x; u[1] = lo.y; u[2] = lo.z; u[3] = lo.w;
  u[4] = hi.x; u[5] = hi.y; u[6] = hi.z; u[7] = hi.w;
  return __builtin_bit_cast(v16bf, u);
}

// Same A fragment built from f32 source (softmax weights), cvt->bf16 in regs.
__device__ __forceinline__ v16bf load_frag_a_f32(const float* tile, int ld) {
  const int lane = threadIdx.x & 31;
  const int r = lane & 15;
  const int c = (lane < 16) ? 0 : 8;
  const float* p = tile + r * ld + c;
  float4 x0 = *(const float4*)(p);
  float4 x1 = *(const float4*)(p + 4);
  float4 y0 = *(const float4*)(p + 16);
  float4 y1 = *(const float4*)(p + 20);
  v8u u;
  u[0] = pack_bf16(x0.x, x0.y); u[1] = pack_bf16(x0.z, x0.w);
  u[2] = pack_bf16(x1.x, x1.y); u[3] = pack_bf16(x1.z, x1.w);
  u[4] = pack_bf16(y0.x, y0.y); u[5] = pack_bf16(y0.z, y0.w);
  u[6] = pack_bf16(y1.x, y1.y); u[7] = pack_bf16(y1.z, y1.w);
  return __builtin_bit_cast(v16bf, u);
}

// B fragment (32x16 bf16). Source S row-major [N,K]; B[k][n] = S[n][k]; tile = &S[n0][k0].
// lane<16: K {0..7,16..23}; lane>=16: K {8..15,24..31}; column = lane&15.
__device__ __forceinline__ v16bf load_frag_b_bf16(const u16* tile, int ld) {
  const int lane = threadIdx.x & 31;
  const int n = lane & 15;
  const int c = (lane < 16) ? 0 : 8;
  const u16* p = tile + n * ld + c;
  uint4 lo = *(const uint4*)(p);
  uint4 hi = *(const uint4*)(p + 16);
  v8u u;
  u[0] = lo.x; u[1] = lo.y; u[2] = lo.z; u[3] = lo.w;
  u[4] = hi.x; u[5] = hi.y; u[6] = hi.z; u[7] = hi.w;
  return __builtin_bit_cast(v16bf, u);
}

__device__ __forceinline__ v8f wmma_bf16(v16bf a, v16bf b, v8f c) {
  return __builtin_amdgcn_wmma_f32_16x16x32_bf16(false, a, false, b, (short)0, c, false, false);
}

// ---------------- kernels ----------------

__global__ void cvt_f32_to_bf16(const float* __restrict__ src, u16* __restrict__ dst, int n) {
  for (int i = blockIdx.x * blockDim.x + threadIdx.x; i < n; i += gridDim.x * blockDim.x)
    dst[i] = f2bf(src[i]);
}

// C[m,n] = A[m,:] . W[n,:] + bias[n]   (A:[4096,1024] bf16, W:[1024,1024] bf16)
// Block = 4 waves, block tile 128(M) x 64(N); per-wave tile 32x64 (8 WMMA / k-step).
// B tile (64 N-rows x 32 K) staged in LDS, double buffered, async copies.
// mode 0: bf16 row-major [M,1024] (Q,K) | mode 1: bf16 transposed -> Vt | mode 2: f32 (final)
#define BPAD 40   // padded LDS row stride in u16 (80B, 16B aligned, bank-skewed)

__global__ __launch_bounds__(128) void gemm_bf16_wmma(
    const u16* __restrict__ A, const u16* __restrict__ W, const float* __restrict__ bias,
    u16* __restrict__ outBf, float* __restrict__ outF, int mode) {
  __shared__ u16 Bsh[2][64 * BPAD];
  const int tid  = threadIdx.x;
  const int wave = tid >> 5;
  const int m0 = blockIdx.x * 128 + wave * 32;
  const int n0 = blockIdx.y * 64;

  // stage W[n0 + r][k0 .. k0+32) into Bsh[buf]; 256 16B chunks over 128 threads
  auto stage = [&](int buf, int k0) {
#pragma unroll
    for (int j = 0; j < 2; ++j) {
      const int id = tid * 2 + j;       // 0..255
      const int r  = id >> 2;           // 0..63
      const int c  = (id & 3) * 8;      // 0,8,16,24 (u16)
      copy16_g2lds(&Bsh[buf][r * BPAD + c], W + (size_t)(n0 + r) * DMODEL + k0 + c);
    }
  };

  v8f acc[2][4] = {};
  const u16* Arow0 = A + (size_t)m0 * DMODEL;
  const u16* Arow1 = A + (size_t)(m0 + 16) * DMODEL;

  stage(0, 0);
  v16bf a0 = load_frag_a_bf16(Arow0, DMODEL);
  v16bf a1 = load_frag_a_bf16(Arow1, DMODEL);
  wait_async_copies();
  __syncthreads();

  int buf = 0;
  for (int k = 0; k < DMODEL; k += 32) {
    const int  kn   = k + 32;
    const bool more = kn < DMODEL;
    if (more) stage(buf ^ 1, kn);              // async: overlaps with compute below
    v16bf na0 = a0, na1 = a1;
    if (more) {                                 // register double-buffer for A
      na0 = load_frag_a_bf16(Arow0 + kn, DMODEL);
      na1 = load_frag_a_bf16(Arow1 + kn, DMODEL);
    }
#pragma unroll
    for (int i = 0; i < 4; ++i) {
      v16bf b = load_frag_b_bf16(&Bsh[buf][16 * i * BPAD], BPAD);   // ds_load_b128
      acc[0][i] = wmma_bf16(a0, b, acc[0][i]);
      acc[1][i] = wmma_bf16(a1, b, acc[1][i]);
    }
    wait_async_copies();
    __syncthreads();
    buf ^= 1; a0 = na0; a1 = na1;
  }

  const int lane = threadIdx.x & 31;
  const int col = lane & 15;
  const int rbase = (lane < 16) ? 0 : 8;
#pragma unroll
  for (int w = 0; w < 2; ++w)
#pragma unroll
    for (int i = 0; i < 4; ++i) {
      const int n = n0 + 16 * i + col;
      const float bv = bias[n];
#pragma unroll
      for (int v = 0; v < 8; ++v) {
        const int m = m0 + 16 * w + rbase + v;
        const float val = acc[w][i][v] + bv;
        if (mode == 0) {
          outBf[(size_t)m * DMODEL + n] = f2bf(val);
        } else if (mode == 1) {
          const int b = m >> 11, t = m & (SEQ - 1);
          outBf[((size_t)(b * DMODEL + n)) * SEQ + t] = f2bf(val);
        } else {
          outF[(size_t)m * DMODEL + n] = val;
        }
      }
    }
}

// scores[bh, t0.., s0..] = Q.K^T / 8, causal strips only. K-dim = 64 = exactly 2 WMMA steps:
// issue all 10 fragment loads in one clause, then 8 back-to-back WMMAs.
// grid: x = sStrip(32), y = tTile(128), z = bh(32); block = 32 (one wave)
__global__ void scores_wmma(const u16* __restrict__ Qb, const u16* __restrict__ Kb,
                            float* __restrict__ wts) {
  const int s0 = blockIdx.x * 64;
  const int t0 = blockIdx.y * 16;
  const int bh = blockIdx.z;
  if (s0 > t0 + 15) return;                    // above diagonal -> softmax writes zeros
  const int b = bh >> 4, h = bh & 15;
  const u16* Qt = Qb + ((size_t)(b * SEQ + t0)) * DMODEL + h * DK;

  v16bf a0 = load_frag_a_bf16(Qt, DMODEL);
  v16bf a1 = load_frag_a_bf16(Qt + 32, DMODEL);
  v16bf b0[4], b1[4];
#pragma unroll
  for (int i = 0; i < 4; ++i) {
    const u16* Ks = Kb + ((size_t)(b * SEQ + s0 + 16 * i)) * DMODEL + h * DK;
    b0[i] = load_frag_b_bf16(Ks, DMODEL);
    b1[i] = load_frag_b_bf16(Ks + 32, DMODEL);
  }
  v8f acc[4] = {};
#pragma unroll
  for (int i = 0; i < 4; ++i) {
    acc[i] = wmma_bf16(a0, b0[i], acc[i]);
    acc[i] = wmma_bf16(a1, b1[i], acc[i]);
  }

  const int lane = threadIdx.x & 31;
  const int col = lane & 15;
  const int rbase = (lane < 16) ? 0 : 8;
  float* wrow = wts + ((size_t)bh * SEQ + t0) * SEQ;
#pragma unroll
  for (int i = 0; i < 4; ++i)
#pragma unroll
    for (int v = 0; v < 8; ++v)
      wrow[(size_t)(rbase + v) * SEQ + s0 + 16 * i + col] = acc[i][v] * 0.125f;
}

// Row-wise causal softmax in place. One 256-thread block per row.
__global__ void softmax_rows(float* __restrict__ wts) {
  const int row = blockIdx.x;                  // bh*2048 + t
  const int t = row & (SEQ - 1);
  float* w = wts + (size_t)row * SEQ;
  __shared__ float red[256];
  const int tid = threadIdx.x;
  float vals[8];
  float mx = -INFINITY;
#pragma unroll
  for (int i = 0; i < 8; ++i) {
    const int j = tid + i * 256;
    const float s = (j <= t) ? w[j] : -INFINITY;
    vals[i] = s;
    mx = fmaxf(mx, s);
  }
  red[tid] = mx; __syncthreads();
  for (int off = 128; off; off >>= 1) {
    if (tid < off) red[tid] = fmaxf(red[tid], red[tid + off]);
    __syncthreads();
  }
  mx = red[0]; __syncthreads();
  float sum = 0.f;
#pragma unroll
  for (int i = 0; i < 8; ++i) {
    const float e = (vals[i] == -INFINITY) ? 0.f : __expf(vals[i] - mx);
    vals[i] = e;
    sum += e;
  }
  red[tid] = sum; __syncthreads();
  for (int off = 128; off; off >>= 1) {
    if (tid < off) red[tid] += red[tid + off];
    __syncthreads();
  }
  const float inv = 1.0f / red[0];
#pragma unroll
  for (int i = 0; i < 8; ++i)
    w[tid + i * 256] = vals[i] * inv;
}

// attn[t,d] = sum_s P[t,s] V[s,d] per (b,h); Vt is [B][1024][T] bf16.
// Software-pipelined k-loop: prefetch next A(f32->bf16) + 4 B frags while computing current.
// grid: x = tTile(128), y = bh(32); block = 32
__global__ void attn_out_wmma(const float* __restrict__ wts, const u16* __restrict__ Vt,
                              u16* __restrict__ attnB) {
  const int t0 = blockIdx.x * 16;
  const int bh = blockIdx.y;
  const int b = bh >> 4, h = bh & 15;
  const float* Prow = wts + ((size_t)bh * SEQ + t0) * SEQ;
  const u16* Vbase = Vt + ((size_t)(b * DMODEL + h * DK)) * SEQ;
  const int kEnd = (t0 + 16 + 31) & ~31;       // causal band rounded to K=32; tail reads zeros

  v16bf a = load_frag_a_f32(Prow, SEQ);
  v16bf bb[4];
#pragma unroll
  for (int i = 0; i < 4; ++i) bb[i] = load_frag_b_bf16(Vbase + (size_t)(16 * i) * SEQ, SEQ);

  v8f acc[4] = {};
  for (int k = 0; k < kEnd; k += 32) {
    const int  kn   = k + 32;
    const bool more = kn < kEnd;
    v16bf na = a, nb0 = bb[0], nb1 = bb[1], nb2 = bb[2], nb3 = bb[3];
    if (more) {
      na  = load_frag_a_f32(Prow + kn, SEQ);
      nb0 = load_frag_b_bf16(Vbase + 0 * SEQ + kn, SEQ);
      nb1 = load_frag_b_bf16(Vbase + 16 * SEQ + kn, SEQ);
      nb2 = load_frag_b_bf16(Vbase + 32 * SEQ + kn, SEQ);
      nb3 = load_frag_b_bf16(Vbase + 48 * SEQ + kn, SEQ);
    }
#pragma unroll
    for (int i = 0; i < 4; ++i) acc[i] = wmma_bf16(a, bb[i], acc[i]);
    a = na; bb[0] = nb0; bb[1] = nb1; bb[2] = nb2; bb[3] = nb3;
  }

  const int lane = threadIdx.x & 31;
  const int col = lane & 15;
  const int rbase = (lane < 16) ? 0 : 8;
#pragma unroll
  for (int i = 0; i < 4; ++i)
#pragma unroll
    for (int v = 0; v < 8; ++v)
      attnB[((size_t)(b * SEQ + t0 + rbase + v)) * DMODEL + h * DK + 16 * i + col] =
          f2bf(acc[i][v]);
}

// ---------------- launch ----------------

extern "C" void kernel_launch(void* const* d_in, const int* in_sizes, int n_in,
                              void* d_out, int out_size, void* d_ws, size_t ws_size,
                              hipStream_t stream) {
  const float* x  = (const float*)d_in[0];
  const float* Wq = (const float*)d_in[1];
  const float* bq = (const float*)d_in[2];
  const float* Wk = (const float*)d_in[3];
  const float* bk = (const float*)d_in[4];
  const float* Wv = (const float*)d_in[5];
  const float* bv = (const float*)d_in[6];
  const float* Wo = (const float*)d_in[7];
  const float* bo = (const float*)d_in[8];

  float* outF = (float*)d_out;                                   // [B,T,D]
  float* wts  = outF + (size_t)BATCH * SEQ * DMODEL;             // [B,H,T,T]

  u16* ws = (u16*)d_ws;
  const size_t NX = (size_t)MROWS * DMODEL;    // 4M u16
  const size_t NW = (size_t)DMODEL * DMODEL;   // 1M u16
  u16* xB   = ws;
  u16* WqB  = xB  + NX;
  u16* WkB  = WqB + NW;
  u16* WvB  = WkB + NW;
  u16* WoB  = WvB + NW;
  u16* QB   = WoB + NW;
  u16* KB   = QB  + NX;
  u16* VtB  = KB  + NX;   // [B][1024][2048]
  u16* attn = VtB + NX;

  cvt_f32_to_bf16<<<2048, 256, 0, stream>>>(x,  xB,  (int)NX);
  cvt_f32_to_bf16<<<1024, 256, 0, stream>>>(Wq, WqB, (int)NW);
  cvt_f32_to_bf16<<<1024, 256, 0, stream>>>(Wk, WkB, (int)NW);
  cvt_f32_to_bf16<<<1024, 256, 0, stream>>>(Wv, WvB, (int)NW);
  cvt_f32_to_bf16<<<1024, 256, 0, stream>>>(Wo, WoB, (int)NW);

  dim3 gGemm(MROWS / 128, DMODEL / 64);
  gemm_bf16_wmma<<<gGemm, 128, 0, stream>>>(xB, WqB, bq, QB,  nullptr, 0);
  gemm_bf16_wmma<<<gGemm, 128, 0, stream>>>(xB, WkB, bk, KB,  nullptr, 0);
  gemm_bf16_wmma<<<gGemm, 128, 0, stream>>>(xB, WvB, bv, VtB, nullptr, 1);

  scores_wmma<<<dim3(SEQ / 64, SEQ / 16, BATCH * HEADS), 32, 0, stream>>>(QB, KB, wts);
  softmax_rows<<<BATCH * HEADS * SEQ, 256, 0, stream>>>(wts);
  attn_out_wmma<<<dim3(SEQ / 16, BATCH * HEADS), 32, 0, stream>>>(wts, VtB, attn);

  gemm_bf16_wmma<<<gGemm, 128, 0, stream>>>(attn, WoB, bo, nullptr, outF, 2);
}
